// Seq2Seq_15126874816988
// MI455X (gfx1250) — matrix operand order repaired and checked by reading, not compiled, not run on hardware
//
#include <hip/hip_runtime.h>

#define HID    512
#define TIN    168
#define TOUT   24
#define FEAT   8
#define BATCH  1024
#define KAUG   544          // 512 + 32 (x padded)
#define KTILES 17           // KAUG / 32
#define NTILES 128          // 2048 / 16
#define HSTRIDE 560         // padded LDS row stride (elements)
#define NFRAG  (KTILES * 16)   // 272 B fragments per wave per step

#define AS1 __attribute__((address_space(1)))
typedef unsigned short AS1 gus;
typedef float          AS1 gfl;

typedef __attribute__((ext_vector_type(16))) __bf16       v16bf;
typedef __attribute__((ext_vector_type(8)))  float        v8f;
typedef __attribute__((ext_vector_type(4)))  unsigned int u32x4;

union AFrag { v16bf v; u32x4 q[2]; };
union BFrag { v16bf v; u32x4 q[2]; };

__device__ __forceinline__ unsigned short f2bf(float f) {
    unsigned u = __float_as_uint(f);
    u += 0x7FFFu + ((u >> 16) & 1u);          // round-to-nearest-even
    return (unsigned short)(u >> 16);
}
__device__ __forceinline__ float bf2f(unsigned short h) {
    return __uint_as_float(((unsigned)h) << 16);
}
__device__ __forceinline__ float fsigmoid(float x) {
    return 1.0f / (1.0f + __expf(-x));
}
__device__ __forceinline__ float ftanh(float x) {
    float e2 = __expf(2.0f * x);
    return (e2 - 1.0f) / (e2 + 1.0f);
}
__device__ __forceinline__ v8f zero8() {
    v8f z = {0.f,0.f,0.f,0.f,0.f,0.f,0.f,0.f};
    return z;
}

// ---------------------------------------------------------------------------
// Pack [Whh | Wih | 0] (2048 x 544) into WMMA-B-fragment-ordered bf16.
// Fragment (kt, ntile): 32 lanes x 16 bf16; lane L holds
//   K = kt*32 + (L/16)*16 + e   (e = 0..15),   N = ntile*16 + (L%16)
// ---------------------------------------------------------------------------
__global__ __launch_bounds__(256)
void pack_weights(const float* __restrict__ Whh, const float* __restrict__ Wih,
                  int Fin, unsigned short* __restrict__ out)
{
    int id = blockIdx.x * 256 + threadIdx.x;      // 17*128*512 total elements
    int e     =  id        & 15;
    int lane  = (id >> 4)  & 31;
    int ntile = (id >> 9)  & 127;
    int kt    =  id >> 16;
    if (kt >= KTILES) return;
    int K = kt * 32 + (lane >> 4) * 16 + e;
    int N = ntile * 16 + (lane & 15);
    float v = 0.0f;
    if (K < HID)                 v = Whh[N * HID + K];
    else if ((K - HID) < Fin)    v = Wih[N * Fin + (K - HID)];
    out[id] = f2bf(v);
}

__global__ __launch_bounds__(256)
void fuse_bias(const float* __restrict__ eih, const float* __restrict__ ehh,
               const float* __restrict__ dih, const float* __restrict__ dhh,
               float* __restrict__ encB, float* __restrict__ decB)
{
    int id = blockIdx.x * 256 + threadIdx.x;
    if (id < 2048)       encB[id]        = eih[id] + ehh[id];
    else if (id < 4096)  decB[id - 2048] = dih[id - 2048] + dhh[id - 2048];
}

// load B fragment f (= kt*16 + idx, idx = tile*4 + gate) via global addrspace
__device__ __forceinline__ void loadB(BFrag& b, const gus* lbase, int f, int wave)
{
    const int kt  = f >> 4;
    const int idx = f & 15;
    const int nt  = (idx & 3) * 32 + wave * 4 + (idx >> 2);  // gate*32 + hidden tile
    const gus* fp = lbase + ((size_t)(kt * NTILES + nt) << 9);
    b.q[0] = *(const AS1 u32x4*)fp;
    b.q[1] = *(const AS1 u32x4*)(fp + 8);
}

// ---------------------------------------------------------------------------
// One LSTM step for a 16-row batch tile. h' (16 x 544 bf16) is in LDS,
// weights streamed from the L2-resident fragment pack via global_load_b128,
// c lives in registers. 272 WMMAs / wave / step, 3-deep B + 2-deep A pipeline.
// ---------------------------------------------------------------------------
__device__ __forceinline__ void lstm_step(
    const unsigned short* pack,                // fragment-packed bf16 weights
    const float* bias,                         // fused bias (2048)
    unsigned short (*hS)[HSTRIDE],             // h' in LDS
    v8f (&c)[4], int lane, int wave)
{
    // Launder both pointers so LICM cannot hoist the time-invariant loads out
    // of the recurrence loop (and spill); rebuild them as address_space(1) so
    // the stream lowers to global_load_b128 (LOADcnt only, no flat/DScnt tie).
    unsigned long long pv = (unsigned long long)pack;
    unsigned long long bv = (unsigned long long)bias;
    asm volatile("" : "+s"(pv), "+s"(bv));
    const gus* gpack = (const gus*)pv;
    const gfl* gbias = (const gfl*)bv;

    v8f acc[16];
    #pragma unroll
    for (int n = 0; n < 16; ++n) acc[n] = zero8();

    const int row = lane & 15;
    const int sub = lane >> 4;
    const gus* lbase = gpack + lane * 16;

    // 3-deep B-fragment ring + 2-deep A double buffer
    BFrag bb[3];
    loadB(bb[0], lbase, 0, wave);
    loadB(bb[1], lbase, 1, wave);
    loadB(bb[2], lbase, 2, wave);

    AFrag aa[2];
    aa[0].q[0] = *(const u32x4*)&hS[row][sub * 16];
    aa[0].q[1] = *(const u32x4*)&hS[row][sub * 16 + 8];

    #pragma unroll
    for (int f = 0; f < NFRAG; ++f) {
        const int kt  = f >> 4;
        const int idx = f & 15;
        acc[idx] = __builtin_amdgcn_wmma_f32_16x16x32_bf16(
            false, aa[kt & 1].v, false, bb[f % 3].v, (short)0, acc[idx], false, false);
        if (f + 3 < NFRAG)
            loadB(bb[f % 3], lbase, f + 3, wave);
        if (idx == 12 && kt + 1 < KTILES) {      // prefetch next kt's A mid-tile
            aa[(kt + 1) & 1].q[0] = *(const u32x4*)&hS[row][(kt + 1) * 32 + sub * 16];
            aa[(kt + 1) & 1].q[1] = *(const u32x4*)&hS[row][(kt + 1) * 32 + sub * 16 + 8];
        }
    }

    __syncthreads();   // all waves done reading h' before overwriting it

    #pragma unroll
    for (int i = 0; i < 4; ++i) {
        const int colh = (wave * 4 + i) * 16 + (lane & 15);
        const float bI = gbias[          colh];
        const float bF = gbias[ 512 +    colh];
        const float bG = gbias[1024 +    colh];
        const float bO = gbias[1536 +    colh];
        #pragma unroll
        for (int e = 0; e < 8; ++e) {
            float si = fsigmoid(acc[i * 4 + 0][e] + bI);
            float sf = fsigmoid(acc[i * 4 + 1][e] + bF);
            float tg = ftanh   (acc[i * 4 + 2][e] + bG);
            float so = fsigmoid(acc[i * 4 + 3][e] + bO);
            float cn = sf * c[i][e] + si * tg;
            c[i][e] = cn;
            float hn = so * ftanh(cn);
            // C/D layout: M = e + 8*(lane/16), N = lane%16
            hS[e + (sub << 3)][colh] = f2bf(hn);
        }
    }
}

// ---------------------------------------------------------------------------
// Persistent per-batch-tile kernel: 64 WGs x 16 rows; full 168-step encoder
// + 24-step decoder + FC head, no inter-block sync required.
// ---------------------------------------------------------------------------
__global__ __launch_bounds__(256)
void seq2seq_kernel(const float* __restrict__ X,
                    const unsigned short* __restrict__ encP,
                    const unsigned short* __restrict__ decP,
                    const float* __restrict__ encB,
                    const float* __restrict__ decB,
                    const float* __restrict__ fcW,
                    const float* __restrict__ fcb,
                    float* __restrict__ out)
{
    __shared__ __align__(16) unsigned short hS[16][HSTRIDE];
    __shared__ float yS[16];

    const int tid  = threadIdx.x;
    const int lane = tid & 31;
    const int wave = tid >> 5;
    const int b0   = blockIdx.x * 16;

    // zero h' (covers h=0, x pad, alignment pad)
    for (int i = tid; i < 16 * HSTRIDE; i += 256)
        ((unsigned short*)hS)[i] = 0;

    v8f c[4];
    #pragma unroll
    for (int i = 0; i < 4; ++i) c[i] = zero8();

    __syncthreads();

    // ---------------- encoder ----------------
    for (int t = 0; t < TIN; ++t) {
        if (tid < 128) {                       // stage x_t into aug cols 512..519
            int r = tid >> 3, f = tid & 7;
            hS[r][HID + f] = f2bf(X[(size_t)(b0 + r) * TIN * FEAT + t * FEAT + f]);
        }
        __syncthreads();
        lstm_step(encP, encB, hS, c, lane, wave);
    }

    // ---------------- decoder init ----------------
    __syncthreads();                           // h writes of last enc step done
    if (tid < 16)  yS[tid] = X[(size_t)(b0 + tid) * TIN * FEAT + (TIN * FEAT - 1)];
    if (tid < 112) {                           // zero stale x features 513..519
        int r = tid / 7, f = 513 + tid % 7;
        hS[r][f] = 0;
    }
    __syncthreads();

    // ---------------- decoder ----------------
    for (int t = 0; t < TOUT; ++t) {
        if (tid < 16) hS[tid][HID] = f2bf(yS[tid]);   // scalar input
        __syncthreads();
        lstm_step(decP, decB, hS, c, lane, wave);
        __syncthreads();                              // h_new visible

        // y = h . fcW + fcb   (16 rows, 16 lanes/row, 32 elems/lane)
        {
            int r = tid >> 4, s = tid & 15;
            float p = 0.0f;
            #pragma unroll
            for (int j = 0; j < 32; ++j)
                p += bf2f(hS[r][s * 32 + j]) * fcW[s * 32 + j];
            p += __shfl_xor(p, 8, 16);
            p += __shfl_xor(p, 4, 16);
            p += __shfl_xor(p, 2, 16);
            p += __shfl_xor(p, 1, 16);
            if (s == 0) {
                float y = p + fcb[0];
                yS[r] = y;
                out[(size_t)(b0 + r) * TOUT + t] = y;
            }
        }
        __syncthreads();                              // yS ready before restage
    }
}

// ---------------------------------------------------------------------------
extern "C" void kernel_launch(void* const* d_in, const int* in_sizes, int n_in,
                              void* d_out, int out_size, void* d_ws, size_t ws_size,
                              hipStream_t stream) {
    (void)in_sizes; (void)n_in; (void)out_size; (void)ws_size;

    const float* X        = (const float*)d_in[0];
    const float* enc_Wih  = (const float*)d_in[1];
    const float* enc_Whh  = (const float*)d_in[2];
    const float* enc_bih  = (const float*)d_in[3];
    const float* enc_bhh  = (const float*)d_in[4];
    const float* dec_Wih  = (const float*)d_in[5];
    const float* dec_Whh  = (const float*)d_in[6];
    const float* dec_bih  = (const float*)d_in[7];
    const float* dec_bhh  = (const float*)d_in[8];
    const float* fc_W     = (const float*)d_in[9];
    const float* fc_b     = (const float*)d_in[10];

    const size_t packElems = (size_t)KTILES * NTILES * 512;   // 1,114,112 bf16
    unsigned short* encP = (unsigned short*)d_ws;
    unsigned short* decP = encP + packElems;
    float* encB = (float*)((char*)d_ws + 2 * packElems * sizeof(unsigned short));
    float* decB = encB + 2048;

    const int packBlocks = (int)((packElems + 255) / 256);    // 4352
    pack_weights<<<packBlocks, 256, 0, stream>>>(enc_Whh, enc_Wih, FEAT, encP);
    pack_weights<<<packBlocks, 256, 0, stream>>>(dec_Whh, dec_Wih, 1,    decP);
    fuse_bias   <<<16, 256, 0, stream>>>(enc_bih, enc_bhh, dec_bih, dec_bhh, encB, decB);

    seq2seq_kernel<<<BATCH / 16, 256, 0, stream>>>(
        X, encP, decP, encB, decB, fc_W, fc_b, (float*)d_out);
}
